// XEyTransformerLayer_48661979464058
// MI455X (gfx1250) — matrix operand order repaired and checked by reading, not compiled, not run on hardware
//
#include <hip/hip_runtime.h>

// ---------------------------------------------------------------------------
// XEyTransformerLayer forward for MI455X (gfx1250), f16 WMMA compute.
// BS=4, N=256, DX=256, DE=64, DY=256, H=8, DF=32, FFX=1024, FFE=128, FFY=256
// ---------------------------------------------------------------------------

#define BSZ 4
#define NN  256
#define DXX 256
#define DEE 64
#define DYY 256
#define DFF 32
#define FFXX 1024
#define FFEE 128
#define FFYY 256
#define LNEPS 1e-5f
#define BSTRIDE 40   // halves per column in LDS B panels (16B aligned, padded)

typedef __attribute__((ext_vector_type(16))) _Float16 v16h;
typedef __attribute__((ext_vector_type(8)))  _Float16 v8h;
typedef __attribute__((ext_vector_type(8)))  float    v8f;
typedef __attribute__((ext_vector_type(4)))  float    v4f;

// ---------------------------------------------------------------------------
// WMMA fragment helpers (layouts per CDNA5 ISA 7.12.2, wave32).
// A (16x32 f16): lane l -> m=l&15, g=l>>4; VGPR v<4 packs K=g*8+2v,+1;
//                v>=4 packs K=16+g*8+2(v-4),+1  => two contiguous 8-float runs.
// B (32x16 f16): lane l -> n=l&15, g=l>>4; VGPR v<4 packs K=g*8+2v,+1; v>=4 packs K=16+g*8+2(v-4),+1
//                => two 8-half runs (k=8g, 16+8g) when stored [n][k] in LDS.
// C/D (16x16 f32): lane l -> n=l&15, g=l>>4; VGPR v -> m = v + 8g.
// ---------------------------------------------------------------------------

__device__ __forceinline__ v8f wmma_f16(v16h a, v16h b, v8f c) {
  return __builtin_amdgcn_wmma_f32_16x16x32_f16(
      /*neg_a=*/false, a, /*neg_b=*/false, b,
      /*c_mod=*/(short)0, c, /*reuse_a=*/false, /*reuse_b=*/false);
}

// Vectorized A fragment from row-major f32 (lda mult of 4, base 16B aligned).
__device__ __forceinline__ v16h load_a_f32(const float* __restrict__ A, long lda,
                                           int m0, int k0, int lane) {
  const int m = m0 + (lane & 15);
  const int g = lane >> 4;
  const float* r = A + (long)m * lda + k0 + g * 8;
  const v4f x0 = *(const v4f*)(r);
  const v4f x1 = *(const v4f*)(r + 4);
  const v4f x2 = *(const v4f*)(r + 16);
  const v4f x3 = *(const v4f*)(r + 20);
  v16h a;
#pragma unroll
  for (int e = 0; e < 4; ++e) {
    a[e]      = (_Float16)x0[e];
    a[4 + e]  = (_Float16)x1[e];
    a[8 + e]  = (_Float16)x2[e];
    a[12 + e] = (_Float16)x3[e];
  }
  return a;
}

// A-fragment with per-column affine:  a = ye1[c] + (ye2[c]+1) * Y[m][c]
__device__ __forceinline__ v16h load_a_affine(const float* __restrict__ A, long lda,
                                              int k0, int lane,
                                              const float* __restrict__ ye1b,
                                              const float* __restrict__ ye2b) {
  const int m = lane & 15;
  const int g = lane >> 4;
  const int cb = k0 + g * 8;
  const float* r = A + (long)m * lda + cb;
  const v4f x0 = *(const v4f*)(r);
  const v4f x1 = *(const v4f*)(r + 4);
  const v4f x2 = *(const v4f*)(r + 16);
  const v4f x3 = *(const v4f*)(r + 20);
  const v4f p0 = *(const v4f*)(ye1b + cb);
  const v4f p1 = *(const v4f*)(ye1b + cb + 4);
  const v4f p2 = *(const v4f*)(ye1b + cb + 16);
  const v4f p3 = *(const v4f*)(ye1b + cb + 20);
  const v4f q0 = *(const v4f*)(ye2b + cb);
  const v4f q1 = *(const v4f*)(ye2b + cb + 4);
  const v4f q2 = *(const v4f*)(ye2b + cb + 16);
  const v4f q3 = *(const v4f*)(ye2b + cb + 20);
  v16h a;
#pragma unroll
  for (int e = 0; e < 4; ++e) {
    a[e]      = (_Float16)(p0[e] + (q0[e] + 1.0f) * x0[e]);
    a[4 + e]  = (_Float16)(p1[e] + (q1[e] + 1.0f) * x1[e]);
    a[8 + e]  = (_Float16)(p2[e] + (q2[e] + 1.0f) * x2[e]);
    a[12 + e] = (_Float16)(p3[e] + (q3[e] + 1.0f) * x3[e]);
  }
  return a;
}

// Cooperative fill of one 32-k B panel into LDS as f16 [ncols][BSTRIDE],
// column-major in k. 128 threads, coalesced float4 global reads.
__device__ __forceinline__ void fill_b_panel(_Float16* __restrict__ lds,
                                             const float* __restrict__ W, long ldw,
                                             int k0, int n0, int ncols, int tid) {
  const int tpr = ncols >> 2;   // threads per row (float4 each)
  const int n4 = (tid % tpr) * 4;
  for (int kk = tid / tpr; kk < 32; kk += 128 / tpr) {
    const v4f w4 = *(const v4f*)&W[(long)(k0 + kk) * ldw + n0 + n4];
    lds[(n4 + 0) * BSTRIDE + kk] = (_Float16)w4[0];
    lds[(n4 + 1) * BSTRIDE + kk] = (_Float16)w4[1];
    lds[(n4 + 2) * BSTRIDE + kk] = (_Float16)w4[2];
    lds[(n4 + 3) * BSTRIDE + kk] = (_Float16)w4[3];
  }
}

// B fragment: two aligned 16B LDS reads per lane.
__device__ __forceinline__ v16h load_b_lds(const _Float16* __restrict__ lds,
                                           int t, int lane) {
  const int n = t * 16 + (lane & 15);
  const int g = lane >> 4;
  const _Float16* p = lds + n * BSTRIDE + g * 8;
  const v8h lo = *(const v8h*)(p);
  const v8h hi = *(const v8h*)(p + 16);
  v16h b;
#pragma unroll
  for (int e = 0; e < 8; ++e) { b[e] = lo[e]; b[8 + e] = hi[e]; }
  return b;
}

// ---------------------------------------------------------------------------
// Generic GEMM: C[M,N] = act(A[M,K] @ W[K,N] + bias) * rowscale[row]
// block = 128 threads (4 waves) covering 64 rows x 64 cols; waves split rows
// and share the LDS-staged B panel. M%64==0, N%64==0, K%32==0.
// ---------------------------------------------------------------------------
__global__ __launch_bounds__(128) void gemm_kernel(
    const float* __restrict__ A, long lda,
    const float* __restrict__ W, long ldw,
    const float* __restrict__ bias,
    const float* __restrict__ rowscale,
    float* __restrict__ C, long ldc,
    int Kd, int relu) {
  __shared__ __align__(16) _Float16 lbs[64 * BSTRIDE];
  const int tid = threadIdx.x;
  const int lane = tid & 31;
  const int wave = tid >> 5;
  const int m0 = (blockIdx.x * 4 + wave) * 16;
  const int n0 = blockIdx.y * 64;
  v8f acc[4] = {v8f{}, v8f{}, v8f{}, v8f{}};
  for (int k0 = 0; k0 < Kd; k0 += 32) {
    fill_b_panel(lbs, W, ldw, k0, n0, 64, tid);
    __syncthreads();
    v16h a = load_a_f32(A, lda, m0, k0, lane);
#pragma unroll
    for (int t = 0; t < 4; ++t)
      acc[t] = wmma_f16(a, load_b_lds(lbs, t, lane), acc[t]);
    __syncthreads();
  }
  const int nlo = lane & 15, g = lane >> 4;
#pragma unroll
  for (int t = 0; t < 4; ++t) {
    const int n = n0 + t * 16 + nlo;
    const float bi = bias ? bias[n] : 0.0f;
#pragma unroll
    for (int v = 0; v < 8; ++v) {
      const int m = m0 + v + 8 * g;
      float val = acc[t][v] + bi;
      if (relu) val = fmaxf(val, 0.0f);
      if (rowscale) val *= rowscale[m];
      C[(long)m * ldc + n] = val;
    }
  }
}

// ---------------------------------------------------------------------------
// Stage B: E1/E2 GEMMs fused with Y epilogue.
// Y[b,i,j,c] = Q[b,j,c]*K[b,i,c]/DF * (E1[b,i,j,c]+1) + E2[b,i,j,c]
// block tile: 64 j-rows x 64 c-cols for one (b,i); waves split j, share
// LDS-staged Wmul/Wadd panels. Grid = 4*256*4*4 = 16384 blocks.
// ---------------------------------------------------------------------------
__global__ __launch_bounds__(128) void stageB_kernel(
    const float* __restrict__ E, const float* __restrict__ Q,
    const float* __restrict__ Km,
    const float* __restrict__ Wmul, const float* __restrict__ bmul,
    const float* __restrict__ Wadd, const float* __restrict__ badd,
    const float* __restrict__ rs_e,
    float* __restrict__ Y) {
  __shared__ __align__(16) _Float16 lbm[64 * BSTRIDE];
  __shared__ __align__(16) _Float16 lba[64 * BSTRIDE];
  const int tid = threadIdx.x;
  const int lane = tid & 31;
  const int wave = tid >> 5;
  const int blk = blockIdx.x;
  const int c0 = (blk & 3) * 64;
  const int j0 = ((blk >> 2) & 3) * 64 + wave * 16;
  const int i = (blk >> 4) & (NN - 1);
  const int b = blk >> 12;
  const float* Eb = E + (long)(b * NN + i) * NN * DEE;  // rows j, ld=64
  v8f am[4] = {v8f{}, v8f{}, v8f{}, v8f{}};
  v8f aa[4] = {v8f{}, v8f{}, v8f{}, v8f{}};
  for (int k0 = 0; k0 < DEE; k0 += 32) {
    fill_b_panel(lbm, Wmul, DXX, k0, c0, 64, tid);
    fill_b_panel(lba, Wadd, DXX, k0, c0, 64, tid);
    __syncthreads();
    v16h a = load_a_f32(Eb, DEE, j0, k0, lane);
#pragma unroll
    for (int t = 0; t < 4; ++t) {
      am[t] = wmma_f16(a, load_b_lds(lbm, t, lane), am[t]);
      aa[t] = wmma_f16(a, load_b_lds(lba, t, lane), aa[t]);
    }
    __syncthreads();
  }
  const int nlo = lane & 15, g = lane >> 4;
  float ms[8];
#pragma unroll
  for (int v = 0; v < 8; ++v)
    ms[v] = rs_e[(long)(b * NN + i) * NN + j0 + v + 8 * g];
#pragma unroll
  for (int t = 0; t < 4; ++t) {
    const int c = c0 + t * 16 + nlo;
    const float kv = Km[(long)(b * NN + i) * DXX + c];   // axis-1 uses K-proj
    const float bm = bmul[c], ba = badd[c];
#pragma unroll
    for (int v = 0; v < 8; ++v) {
      const int j = j0 + v + 8 * g;
      const float e1 = (am[t][v] + bm) * ms[v];
      const float e2 = (aa[t][v] + ba) * ms[v];
      const float qv = Q[(long)(b * NN + j) * DXX + c];  // axis-2 uses Q-proj
      const float yv = qv * kv * (1.0f / DFF) * (e1 + 1.0f) + e2;
      Y[((long)(b * NN + i) * NN + j) * DXX + c] = yv;
    }
  }
}

// ---------------------------------------------------------------------------
// Stage C: newE = (ye1 + (ye2+1)*Y) @ W_e_out + b  (affine folded into A-load),
// * e_mask, + E residual, LayerNorm(normE1) -> Er.
// block = 128 threads / 64 rows; wave = 16 rows x 64 cols; grid = 262144/64.
// ---------------------------------------------------------------------------
__global__ __launch_bounds__(128) void stageC_kernel(
    const float* __restrict__ Y,
    const float* __restrict__ Weout, const float* __restrict__ beout,
    const float* __restrict__ ye1, const float* __restrict__ ye2,
    const float* __restrict__ rs_e, const float* __restrict__ E,
    const float* __restrict__ gam, const float* __restrict__ bet,
    float* __restrict__ Er) {
  __shared__ __align__(16) _Float16 lbs[64 * BSTRIDE];
  __shared__ float tile[64][DEE + 1];
  __shared__ float mean_s[64], rstd_s[64];
  const int tid = threadIdx.x;
  const int lane = tid & 31;
  const int wave = tid >> 5;
  const long r0 = (long)blockIdx.x * 64;
  const long rw = r0 + wave * 16;
  const int b = (int)(rw >> 16);  // rows per batch = N*N = 65536
  const float* ye1b = ye1 + b * DYY;
  const float* ye2b = ye2 + b * DYY;
  v8f acc[4] = {v8f{}, v8f{}, v8f{}, v8f{}};
  for (int k0 = 0; k0 < DXX; k0 += 32) {
    fill_b_panel(lbs, Weout, DEE, k0, 0, 64, tid);
    __syncthreads();
    v16h a = load_a_affine(Y + rw * DXX, DXX, k0, lane, ye1b, ye2b);
#pragma unroll
    for (int t = 0; t < 4; ++t)
      acc[t] = wmma_f16(a, load_b_lds(lbs, t, lane), acc[t]);
    __syncthreads();
  }
  const int nlo = lane & 15, g = lane >> 4;
#pragma unroll
  for (int t = 0; t < 4; ++t) {
    const int n = t * 16 + nlo;
    const float bi = beout[n];
#pragma unroll
    for (int v = 0; v < 8; ++v) {
      const long r = rw + v + 8 * g;
      tile[wave * 16 + v + 8 * g][n] = (acc[t][v] + bi) * rs_e[r] + E[r * DEE + n];
    }
  }
  __syncthreads();
  if (tid < 64) {
    float s = 0.f, sq = 0.f;
#pragma unroll
    for (int c = 0; c < DEE; ++c) { float v = tile[tid][c]; s += v; sq += v * v; }
    const float mean = s * (1.0f / DEE);
    const float var = sq * (1.0f / DEE) - mean * mean;
    mean_s[tid] = mean;
    rstd_s[tid] = rsqrtf(var + LNEPS);
  }
  __syncthreads();
  for (int q = 0; q < 32; ++q) {
    const int e = q * 128 + tid;
    const int row = e >> 6, col = e & 63;
    const float val = (tile[row][col] - mean_s[row]) * rstd_s[row] * gam[col] + bet[col];
    Er[(r0 + row) * DEE + col] = val;
  }
}

// ---------------------------------------------------------------------------
// Stage D: per-(b,i) online softmax over j (masked) and weighted sum with V.
// grid = BS*N blocks, 256 threads (thread = channel c).
// ---------------------------------------------------------------------------
__global__ __launch_bounds__(256) void attn_kernel(
    const float* __restrict__ Y, const float* __restrict__ V,
    const unsigned char* __restrict__ nm, float* __restrict__ wV) {
  const int bi = blockIdx.x;
  const int b = bi >> 8;
  const int c = threadIdx.x;
  const float* Yp = Y + (long)bi * NN * DXX + c;
  const float* Vp = V + (long)b * NN * DXX + c;
  float m = -1e30f, s = 0.f, acc = 0.f;
  for (int j = 0; j < NN; ++j) {
    if (nm[b * NN + j]) {
      const float yv = Yp[(long)j * DXX];
      const float mn = fmaxf(m, yv);
      const float corr = __expf(m - mn);
      const float w = __expf(yv - mn);
      s = s * corr + w;
      acc = acc * corr + w * Vp[(long)j * DXX];
      m = mn;
    }
  }
  wV[(long)bi * DXX + c] = acc / s;
}

// ---------------------------------------------------------------------------
// Stage F: Eo = LN(Er + linE2(relu(linE1(Er))), normE2) -> d_out E region.
// wave = 16 rows; hidden 16x128 re-staged through LDS (stride 132, aligned).
// ---------------------------------------------------------------------------
__global__ __launch_bounds__(128) void stageF_kernel(
    const float* __restrict__ Er,
    const float* __restrict__ W1, const float* __restrict__ b1,
    const float* __restrict__ W2, const float* __restrict__ b2,
    const float* __restrict__ gam, const float* __restrict__ bet,
    float* __restrict__ Eo) {
  __shared__ __align__(16) _Float16 lbs[128 * BSTRIDE];
  __shared__ __align__(16) float hid[4][16][FFEE + 4];
  __shared__ float tile[64][DEE + 1];
  __shared__ float mean_s[64], rstd_s[64];
  const int tid = threadIdx.x;
  const int lane = tid & 31;
  const int wave = tid >> 5;
  const long r0 = (long)blockIdx.x * 64;
  const long rw = r0 + wave * 16;
  v8f h[8] = {v8f{}, v8f{}, v8f{}, v8f{}, v8f{}, v8f{}, v8f{}, v8f{}};
  for (int k0 = 0; k0 < DEE; k0 += 32) {
    fill_b_panel(lbs, W1, FFEE, k0, 0, 128, tid);
    __syncthreads();
    v16h a = load_a_f32(Er + rw * DEE, DEE, 0, k0, lane);
#pragma unroll
    for (int t = 0; t < 8; ++t)
      h[t] = wmma_f16(a, load_b_lds(lbs, t, lane), h[t]);
    __syncthreads();
  }
  const int nlo = lane & 15, g = lane >> 4;
#pragma unroll
  for (int t = 0; t < 8; ++t) {
    const int n = t * 16 + nlo;
    const float bi = b1[n];
#pragma unroll
    for (int v = 0; v < 8; ++v)
      hid[wave][v + 8 * g][n] = fmaxf(h[t][v] + bi, 0.0f);
  }
  __syncthreads();
  v8f acc[4] = {v8f{}, v8f{}, v8f{}, v8f{}};
  for (int k0 = 0; k0 < FFEE; k0 += 32) {
    fill_b_panel(lbs, W2, DEE, k0, 0, 64, tid);
    __syncthreads();
    v16h a = load_a_f32(&hid[wave][0][0], FFEE + 4, 0, k0, lane);
#pragma unroll
    for (int t = 0; t < 4; ++t)
      acc[t] = wmma_f16(a, load_b_lds(lbs, t, lane), acc[t]);
    __syncthreads();
  }
#pragma unroll
  for (int t = 0; t < 4; ++t) {
    const int n = t * 16 + nlo;
    const float bi = b2[n];
#pragma unroll
    for (int v = 0; v < 8; ++v) {
      const long r = rw + v + 8 * g;
      tile[wave * 16 + v + 8 * g][n] = acc[t][v] + bi + Er[r * DEE + n];
    }
  }
  __syncthreads();
  if (tid < 64) {
    float s = 0.f, sq = 0.f;
#pragma unroll
    for (int c = 0; c < DEE; ++c) { float v = tile[tid][c]; s += v; sq += v * v; }
    const float mean = s * (1.0f / DEE);
    const float var = sq * (1.0f / DEE) - mean * mean;
    mean_s[tid] = mean;
    rstd_s[tid] = rsqrtf(var + LNEPS);
  }
  __syncthreads();
  for (int q = 0; q < 32; ++q) {
    const int e = q * 128 + tid;
    const int row = e >> 6, col = e & 63;
    const float val = (tile[row][col] - mean_s[row]) * rstd_s[row] * gam[col] + bet[col];
    Eo[(r0 + row) * DEE + col] = val;
  }
}

// ---------------------------------------------------------------------------
// Small helpers
// ---------------------------------------------------------------------------
__global__ void build_masks_kernel(const unsigned char* __restrict__ nm,
                                   float* __restrict__ rs_x,
                                   float* __restrict__ rs_e) {
  const int idx = blockIdx.x * blockDim.x + threadIdx.x;  // 262144 threads
  if (idx < BSZ * NN) rs_x[idx] = nm[idx] ? 1.0f : 0.0f;
  const int j = idx & (NN - 1);
  const int bi = idx >> 8;
  const int b = bi >> 8;
  rs_e[idx] = (nm[bi] && nm[b * NN + j]) ? 1.0f : 0.0f;
}

__global__ void small_dense_kernel(const float* __restrict__ A, int Kd,
                                   const float* __restrict__ W, int Nd,
                                   const float* __restrict__ bias,
                                   const float* __restrict__ addend, int relu,
                                   float* __restrict__ C) {
  const int r = blockIdx.x, o = threadIdx.x;
  if (o >= Nd) return;
  float s = bias[o];
  for (int k = 0; k < Kd; ++k) s += A[r * Kd + k] * W[(long)k * Nd + o];
  if (addend) s += addend[r * Nd + o];
  if (relu) s = fmaxf(s, 0.0f);
  C[r * Nd + o] = s;
}

// out = LN(x + resid) over d=256, one block of 256 per row.
__global__ __launch_bounds__(256) void ln256_kernel(
    const float* __restrict__ x, const float* __restrict__ resid,
    const float* __restrict__ gam, const float* __restrict__ bet,
    float* __restrict__ out) {
  __shared__ float s_sum[256], s_sq[256];
  const long r = blockIdx.x;
  const int t = threadIdx.x;
  float v = x[r * 256 + t] + (resid ? resid[r * 256 + t] : 0.0f);
  s_sum[t] = v;
  s_sq[t] = v * v;
  __syncthreads();
  for (int s = 128; s > 0; s >>= 1) {
    if (t < s) { s_sum[t] += s_sum[t + s]; s_sq[t] += s_sq[t + s]; }
    __syncthreads();
  }
  const float mean = s_sum[0] * (1.0f / 256.0f);
  const float var = s_sq[0] * (1.0f / 256.0f) - mean * mean;
  out[r * 256 + t] = (v - mean) * rsqrtf(var + LNEPS) * gam[t] + bet[t];
}

__global__ void axin_kernel(const float* __restrict__ wV,
                            const float* __restrict__ yx1,
                            const float* __restrict__ yx2,
                            float* __restrict__ Axin) {
  const long idx = (long)blockIdx.x * 256 + threadIdx.x;  // 262144
  const int c = (int)(idx & 255);
  const int b = (int)(idx >> 16);
  Axin[idx] = yx1[b * DYY + c] + (yx2[b * DYY + c] + 1.0f) * wV[idx];
}

__global__ void zx_kernel(const float* __restrict__ X, float* __restrict__ zx) {
  const int b = blockIdx.x, d = threadIdx.x;
  const float* p = X + (long)b * NN * DXX + d;
  float s = 0.f, sq = 0.f, mn = 1e30f, mx = -1e30f;
  for (int i = 0; i < NN; ++i) {
    const float v = p[(long)i * DXX];
    s += v; sq += v * v; mn = fminf(mn, v); mx = fmaxf(mx, v);
  }
  const float mean = s * (1.0f / NN);
  zx[b * 4 * DXX + d] = mean;
  zx[b * 4 * DXX + DXX + d] = mn;
  zx[b * 4 * DXX + 2 * DXX + d] = mx;
  zx[b * 4 * DXX + 3 * DXX + d] = sqrtf(fmaxf(sq * (1.0f / NN) - mean * mean, 0.0f));
}

__global__ void ze_kernel(const float* __restrict__ E, float* __restrict__ ze) {
  __shared__ float ss[256], ssq[256], smn[256], smx[256];
  const int b = blockIdx.x, t = threadIdx.x;
  const int d = t & 63, sl = t >> 6;  // 4 slices of 16384 (i,j) pairs
  const float* p = E + ((long)b * 65536 + (long)sl * 16384) * DEE + d;
  float s = 0.f, sq = 0.f, mn = 1e30f, mx = -1e30f;
  for (int q = 0; q < 16384; ++q) {
    const float v = p[(long)q * DEE];
    s += v; sq += v * v; mn = fminf(mn, v); mx = fmaxf(mx, v);
  }
  ss[t] = s; ssq[t] = sq; smn[t] = mn; smx[t] = mx;
  __syncthreads();
  if (sl == 0) {
    for (int u = 1; u < 4; ++u) {
      s += ss[d + 64 * u]; sq += ssq[d + 64 * u];
      mn = fminf(mn, smn[d + 64 * u]); mx = fmaxf(mx, smx[d + 64 * u]);
    }
    const float mean = s * (1.0f / 65536.0f);
    ze[b * 4 * DEE + d] = mean;
    ze[b * 4 * DEE + DEE + d] = mn;
    ze[b * 4 * DEE + 2 * DEE + d] = mx;
    ze[b * 4 * DEE + 3 * DEE + d] =
        sqrtf(fmaxf(sq * (1.0f / 65536.0f) - mean * mean, 0.0f));
  }
}

// ---------------------------------------------------------------------------
// Host driver
// ---------------------------------------------------------------------------
extern "C" void kernel_launch(void* const* d_in, const int* in_sizes, int n_in,
                              void* d_out, int out_size, void* d_ws, size_t ws_size,
                              hipStream_t stream) {
  (void)in_sizes; (void)n_in; (void)out_size; (void)ws_size;
  const float* X = (const float*)d_in[0];
  const float* E = (const float*)d_in[1];
  const float* y = (const float*)d_in[2];
  const unsigned char* nm = (const unsigned char*)d_in[3];
  // d_in[4] = deterministic (unused)
  // Dense params in dims-dict order: (w,b) pairs starting at index 5.
  const float *Wq = (const float*)d_in[5],  *bq = (const float*)d_in[6];
  const float *Wk = (const float*)d_in[7],  *bk = (const float*)d_in[8];
  const float *Wv = (const float*)d_in[9],  *bv = (const float*)d_in[10];
  const float *We_add = (const float*)d_in[11], *be_add = (const float*)d_in[12];
  const float *We_mul = (const float*)d_in[13], *be_mul = (const float*)d_in[14];
  const float *Wyem = (const float*)d_in[15], *byem = (const float*)d_in[16];
  const float *Wyea = (const float*)d_in[17], *byea = (const float*)d_in[18];
  const float *Wyxm = (const float*)d_in[19], *byxm = (const float*)d_in[20];
  const float *Wyxa = (const float*)d_in[21], *byxa = (const float*)d_in[22];
  const float *Wyy = (const float*)d_in[23], *byy = (const float*)d_in[24];
  const float *Wxy = (const float*)d_in[25], *bxy = (const float*)d_in[26];
  const float *Wey = (const float*)d_in[27], *bey = (const float*)d_in[28];
  const float *Wxo = (const float*)d_in[29], *bxo = (const float*)d_in[30];
  const float *Weo = (const float*)d_in[31], *beo = (const float*)d_in[32];
  const float *Wyo1 = (const float*)d_in[33], *byo1 = (const float*)d_in[34];
  const float *Wyo2 = (const float*)d_in[35], *byo2 = (const float*)d_in[36];
  const float *WlX1 = (const float*)d_in[37], *blX1 = (const float*)d_in[38];
  const float *WlX2 = (const float*)d_in[39], *blX2 = (const float*)d_in[40];
  const float *WlE1 = (const float*)d_in[41], *blE1 = (const float*)d_in[42];
  const float *WlE2 = (const float*)d_in[43], *blE2 = (const float*)d_in[44];
  const float *Wly1 = (const float*)d_in[45], *bly1 = (const float*)d_in[46];
  const float *Wly2 = (const float*)d_in[47], *bly2 = (const float*)d_in[48];
  const float *gX1 = (const float*)d_in[49], *bX1 = (const float*)d_in[50];
  const float *gX2 = (const float*)d_in[51], *bX2 = (const float*)d_in[52];
  const float *gE1 = (const float*)d_in[53], *bE1 = (const float*)d_in[54];
  const float *gE2 = (const float*)d_in[55], *bE2 = (const float*)d_in[56];
  const float *gy1 = (const float*)d_in[57], *by1 = (const float*)d_in[58];
  const float *gy2 = (const float*)d_in[59], *by2 = (const float*)d_in[60];

  float* outX = (float*)d_out;                       // 4*256*256
  float* outE = outX + (long)BSZ * NN * DXX;         // 4*256*256*64
  float* outY = outE + (long)BSZ * NN * NN * DEE;    // 4*256

  // --- workspace carve-up ---
  char* base = (char*)d_ws;
  size_t off = 0;
  auto take = [&](size_t bytes) -> float* {
    off = (off + 255) & ~(size_t)255;
    float* p = (float*)(base + off);
    off += bytes;
    return p;
  };
  const size_t SX = (size_t)BSZ * NN * DXX * 4;        // 1 MiB
  float* Qb   = take(SX);
  float* Kb   = take(SX);
  float* Vb   = take(SX);
  float* rs_x = take((size_t)BSZ * NN * 4);
  float* rs_e = take((size_t)BSZ * NN * NN * 4);
  float* ye1  = take(BSZ * DYY * 4);
  float* ye2  = take(BSZ * DYY * 4);
  float* yx1  = take(BSZ * DYY * 4);
  float* yx2  = take(BSZ * DYY * 4);
  float* Ybuf = take((size_t)BSZ * NN * NN * DXX * 4); // 256 MiB
  float* Erb  = take((size_t)BSZ * NN * NN * DEE * 4); // 64 MiB
  float* wV   = take(SX);
  float* Axin = take(SX);
  float* newX = take(SX);
  float* Xr   = take(SX);
  float* Xh   = take((size_t)BSZ * NN * FFXX * 4);     // 4 MiB
  float* Xff  = take(SX);
  float* zx   = take((size_t)BSZ * 4 * DXX * 4);
  float* ze   = take((size_t)BSZ * 4 * DEE * 4);
  float* yt1  = take(BSZ * DYY * 4);
  float* yt2  = take(BSZ * DYY * 4);
  float* yt3  = take(BSZ * DYY * 4);
  float* yt4  = take(BSZ * DYY * 4);
  float* yt5  = take(BSZ * DYY * 4);
  float* yrb  = take(BSZ * DYY * 4);
  float* yt6  = take(BSZ * DYY * 4);
  float* yt7  = take(BSZ * DYY * 4);

  const dim3 B128(128), B256(256);

  // masks
  build_masks_kernel<<<1024, B256, 0, stream>>>(nm, rs_x, rs_e);

  // y-conditioning vectors (note: ye1 uses y_e_add, ye2 uses y_e_mul, etc.)
  small_dense_kernel<<<BSZ, B256, 0, stream>>>(y, DYY, Wyea, DXX, byea, nullptr, 0, ye1);
  small_dense_kernel<<<BSZ, B256, 0, stream>>>(y, DYY, Wyem, DXX, byem, nullptr, 0, ye2);
  small_dense_kernel<<<BSZ, B256, 0, stream>>>(y, DYY, Wyxa, DXX, byxa, nullptr, 0, yx1);
  small_dense_kernel<<<BSZ, B256, 0, stream>>>(y, DYY, Wyxm, DXX, byxm, nullptr, 0, yx2);

  // Q/K/V (masked rows)
  const dim3 gQKV(BSZ * NN / 64, DXX / 64);
  gemm_kernel<<<gQKV, B128, 0, stream>>>(X, DXX, Wq, DXX, bq, rs_x, Qb, DXX, DXX, 0);
  gemm_kernel<<<gQKV, B128, 0, stream>>>(X, DXX, Wk, DXX, bk, rs_x, Kb, DXX, DXX, 0);
  gemm_kernel<<<gQKV, B128, 0, stream>>>(X, DXX, Wv, DXX, bv, rs_x, Vb, DXX, DXX, 0);

  // Stage B: E1/E2 + Y
  stageB_kernel<<<16384, B128, 0, stream>>>(E, Qb, Kb, We_mul, be_mul, We_add, be_add,
                                            rs_e, Ybuf);

  // Stage C: e_out GEMM + mask + residual + LN(normE1) -> Er
  stageC_kernel<<<BSZ * NN * NN / 64, B128, 0, stream>>>(
      Ybuf, Weo, beo, ye1, ye2, rs_e, E, gE1, bE1, Erb);

  // Stage D: masked online softmax over j + weighted V
  attn_kernel<<<BSZ * NN, B256, 0, stream>>>(Ybuf, Vb, nm, wV);

  // X path
  axin_kernel<<<1024, B256, 0, stream>>>(wV, yx1, yx2, Axin);
  gemm_kernel<<<gQKV, B128, 0, stream>>>(Axin, DXX, Wxo, DXX, bxo, rs_x, newX, DXX, DXX, 0);
  ln256_kernel<<<BSZ * NN, B256, 0, stream>>>(X, newX, gX1, bX1, Xr);
  const dim3 gX1d(BSZ * NN / 64, FFXX / 64);
  gemm_kernel<<<gX1d, B128, 0, stream>>>(Xr, DXX, WlX1, FFXX, blX1, nullptr, Xh, FFXX, DXX, 1);
  gemm_kernel<<<gQKV, B128, 0, stream>>>(Xh, FFXX, WlX2, DXX, blX2, nullptr, Xff, DXX, FFXX, 0);
  ln256_kernel<<<BSZ * NN, B256, 0, stream>>>(Xr, Xff, gX2, bX2, outX);

  // Stage F: ffE + residual + LN(normE2) -> Eo
  stageF_kernel<<<BSZ * NN * NN / 64, B128, 0, stream>>>(
      Erb, WlE1, blE1, WlE2, blE2, gE2, bE2, outE);

  // y path
  zx_kernel<<<BSZ, B256, 0, stream>>>(X, zx);
  ze_kernel<<<BSZ, B256, 0, stream>>>(E, ze);
  small_dense_kernel<<<BSZ, B256, 0, stream>>>(y, DYY, Wyy, DYY, byy, nullptr, 0, yt1);
  small_dense_kernel<<<BSZ, B256, 0, stream>>>(zx, 4 * DXX, Wxy, DYY, bxy, yt1, 0, yt2);
  small_dense_kernel<<<BSZ, B256, 0, stream>>>(ze, 4 * DEE, Wey, DYY, bey, yt2, 0, yt3);
  small_dense_kernel<<<BSZ, B256, 0, stream>>>(yt3, DYY, Wyo1, DYY, byo1, nullptr, 1, yt4);
  small_dense_kernel<<<BSZ, B256, 0, stream>>>(yt4, DYY, Wyo2, DYY, byo2, nullptr, 0, yt5);
  ln256_kernel<<<BSZ, B256, 0, stream>>>(y, yt5, gy1, by1, yrb);
  small_dense_kernel<<<BSZ, B256, 0, stream>>>(yrb, DYY, Wly1, FFYY, bly1, yrb, 1, yt6);
  small_dense_kernel<<<BSZ, B256, 0, stream>>>(yt6, FFYY, Wly2, DYY, bly2, nullptr, 0, yt7);
  ln256_kernel<<<BSZ, B256, 0, stream>>>(yrb, yt7, gy2, by2, outY);
}
